// NPChangeTransitionPrior_9268539425200
// MI455X (gfx1250) — matrix operand order repaired and checked by reading, not compiled, not run on hardware
//
#include <hip/hip_runtime.h>
#include <math.h>

// ---------------------------------------------------------------------------
// NPChangeTransitionPrior on MI455X (gfx1250, wave32, WMMA bf16 16x16x32)
//
// Shapes: B=16, LL=514, D=16, H=128, E=128, LAGS=2, length=512, N=8192
// GIN = 161 = 160 (=5*32 WMMA K-steps) + 1 rank-1 column (x_t)
//
// Pipeline (all on `stream`, graph-capture safe):
//   prep_kernel : f32->bf16 weight conversion into WMMA B-fragment layout,
//                 transposed copies for the backward GEMMs, W0 last row /
//                 W3 column extraction, x-lag columns of `base`, zero logdet
//   fc_mlp      : 4-layer MLP on embeddings -> base[:,0:128] (bf16)
//   g_net       : per-(n-tile, d) block: forward 4 layers + analytic
//                 Jacobian backward, fully LDS-resident, WMMA everywhere
// Workspace use: ~5.6 MB.
// ---------------------------------------------------------------------------

typedef __attribute__((ext_vector_type(16))) __bf16 v16bf;
typedef __attribute__((ext_vector_type(8)))  float  v8f;

union FragU { v16bf v; float4 f4[2]; };

__device__ __forceinline__ v8f vzero8() {
  v8f z;
#pragma unroll
  for (int i = 0; i < 8; ++i) z[i] = 0.f;
  return z;
}

// One wave computes a 16(M) x 64(N) stripe: 4 accumulators of 16x16.
// A: row-major bf16 in LDS (lda elements/row).  B: pre-fragmented in global.
// A-frag (ISA 7.12.2, 16-bit A 16x32): lane<16 row=lane K={0..7,16..23}+ks*32,
// lane>=16 same row, K shifted by 8.  Two 16B chunks -> ds_load_b128 pair.
// B-frag: converter stored [ks][ntile][lane][16] contiguously -> 2x b128.
__device__ __forceinline__ void wmma_stripe(
    const __bf16* __restrict__ A, int lda,
    const __bf16* __restrict__ Bf,
    int ksteps, int mrow, int nh, int lane, v8f acc[4])
{
  const int r     = mrow + (lane & 15);
  const int koff8 = (lane >> 4) * 8;
  for (int ks = 0; ks < ksteps; ++ks) {
    FragU a;
    const __bf16* ap = A + r * lda + ks * 32 + koff8;
    a.f4[0] = *(const float4*)(ap);
    a.f4[1] = *(const float4*)(ap + 16);
#pragma unroll
    for (int nt = 0; nt < 4; ++nt) {
      FragU b;
      const float4* bp =
          (const float4*)(Bf + (((ks * 8) + (nh * 4 + nt)) * 32 + lane) * 16);
      b.f4[0] = bp[0];
      b.f4[1] = bp[1];
      acc[nt] = __builtin_amdgcn_wmma_f32_16x16x32_bf16(
          false, a.v, false, b.v, (short)0, acc[nt], false, false);
    }
  }
}

// ---------------- workspace layout (byte offsets, all 256B aligned) --------
static constexpr size_t OFF_FCW  = 0;        // 4 * 16384 bf16   (128 KB)
static constexpr size_t OFF_G0F  = 131072;   // 16 * 20480 bf16  (640 KB)
static constexpr size_t OFF_G1F  = 786432;   // 16 * 16384 bf16  (512 KB)
static constexpr size_t OFF_G2F  = 1310720;
static constexpr size_t OFF_G1T  = 1835008;
static constexpr size_t OFF_G2T  = 2359296;
static constexpr size_t OFF_W0L  = 2883584;  // 16*128 f32
static constexpr size_t OFF_W3   = 2891776;  // 16*128 f32
static constexpr size_t OFF_BASE = 2899968;  // 8192*160 bf16 (2.5 MB)

// B-fragment index decomposition helper (shared by converter and GEMM):
//   linear e = ((ks*8 + nt)*32 + lane)*16 + j
//   K = ks*32 + (lane>>4)*8 + (j&7) + (j>>3)*16 ,  N = nt*16 + (lane&15)
__device__ __forceinline__ void frag_src(int e, int& K, int& N) {
  int j    = e & 15;
  int lane = (e >> 4) & 31;
  int nt   = (e >> 9) & 7;
  int ks   = e >> 12;
  K = ks * 32 + ((lane >> 4) << 3) + (j & 7) + ((j >> 3) << 4);
  N = nt * 16 + (lane & 15);
}

// ---------------------------------------------------------------------------
__global__ __launch_bounds__(256) void prep_kernel(
    const float* __restrict__ x,
    const float* __restrict__ fcW0, const float* __restrict__ fcW1,
    const float* __restrict__ fcW2, const float* __restrict__ fcW3,
    const float* __restrict__ gW0,  const float* __restrict__ gW1,
    const float* __restrict__ gW2,  const float* __restrict__ gW3,
    __bf16* __restrict__ fcWf, __bf16* __restrict__ g0f,
    __bf16* __restrict__ g1f,  __bf16* __restrict__ g2f,
    __bf16* __restrict__ g1tf, __bf16* __restrict__ g2tf,
    float* __restrict__ w0last, float* __restrict__ w3v,
    __bf16* __restrict__ baseB, float* __restrict__ outLog)
{
  long idx = (long)blockIdx.x * 256 + threadIdx.x;
  if (idx < 65536) {                       // fc weight fragments (4 layers)
    int e = (int)idx;
    int l = e >> 14; int r = e & 16383;
    int K, N; frag_src(r, K, N);
    const float* W = (l == 0) ? fcW0 : (l == 1) ? fcW1 : (l == 2) ? fcW2 : fcW3;
    fcWf[e] = (__bf16)W[K * 128 + N];
  } else if (idx < 393216) {               // g_W0 fragments (K padded view 160)
    int e = (int)(idx - 65536);
    int d = e / 20480; int r = e % 20480;
    int K, N; frag_src(r, K, N);           // ks in 0..4 -> K in 0..159
    g0f[e] = (__bf16)gW0[d * 20608 + K * 128 + N];   // 161*128 = 20608
  } else if (idx < 655360) {               // g_W1 forward fragments
    int e = (int)(idx - 393216);
    int d = e >> 14; int r = e & 16383;
    int K, N; frag_src(r, K, N);
    g1f[e] = (__bf16)gW1[d * 16384 + K * 128 + N];
  } else if (idx < 917504) {               // g_W2 forward fragments
    int e = (int)(idx - 655360);
    int d = e >> 14; int r = e & 16383;
    int K, N; frag_src(r, K, N);
    g2f[e] = (__bf16)gW2[d * 16384 + K * 128 + N];
  } else if (idx < 1179648) {              // g_W1^T fragments (backward)
    int e = (int)(idx - 917504);
    int d = e >> 14; int r = e & 16383;
    int K, N; frag_src(r, K, N);
    g1tf[e] = (__bf16)gW1[d * 16384 + N * 128 + K];
  } else if (idx < 1441792) {              // g_W2^T fragments (backward)
    int e = (int)(idx - 1179648);
    int d = e >> 14; int r = e & 16383;
    int K, N; frag_src(r, K, N);
    g2tf[e] = (__bf16)gW2[d * 16384 + N * 128 + K];
  } else if (idx < 1443840) {              // W0 row 160 (rank-1 / Jacobian)
    int e = (int)(idx - 1441792);
    int d = e >> 7; int c = e & 127;
    w0last[e] = gW0[d * 20608 + 160 * 128 + c];
  } else if (idx < 1445888) {              // W3 column
    int e = (int)(idx - 1443840);
    w3v[e] = gW3[e];
  } else if (idx < 1454080) {              // zero logdet accumulator
    outLog[(int)(idx - 1445888)] = 0.f;
  } else {                                 // base[:,128:160] = x lags (bf16)
    int e = (int)(idx - 1454080);          // 8192*32 entries
    int n = e >> 5; int c = e & 31;
    int b = n >> 9; int t = n & 511;
    int lag = c >> 4; int dd = c & 15;
    baseB[n * 160 + 128 + c] = (__bf16)x[(b * 514 + t + lag) * 16 + dd];
  }
}

// ---------------------------------------------------------------------------
// fc MLP: 64-row tile, 8 waves, ping-pong LDS; emits base[:,0:128] in bf16.
__global__ __launch_bounds__(256) void fc_mlp_kernel(
    const float* __restrict__ emb, const __bf16* __restrict__ fcWf,
    const float* __restrict__ b0, const float* __restrict__ b1,
    const float* __restrict__ b2, const float* __restrict__ b3,
    __bf16* __restrict__ baseB)
{
  __shared__ __attribute__((aligned(16))) __bf16 sA[64 * 136];
  __shared__ __attribute__((aligned(16))) __bf16 sB[64 * 136];
  const int tid = threadIdx.x;
  const int n0  = blockIdx.x * 64;

  // stage embeddings tile (rows are positions t+LAGS of each sequence)
#pragma unroll
  for (int i = 0; i < 8; ++i) {
    int idx = tid + i * 256;               // 0..2047 float4s
    int row = idx >> 5; int c4 = idx & 31;
    int n = n0 + row; int b = n >> 9; int t = n & 511;
    float4 v = *(const float4*)(emb + (long)(b * 514 + t + 2) * 128 + c4 * 4);
    __bf16* dp = sA + row * 136 + c4 * 4;
    dp[0] = (__bf16)v.x; dp[1] = (__bf16)v.y;
    dp[2] = (__bf16)v.z; dp[3] = (__bf16)v.w;
  }
  __syncthreads();

  const int lane = tid & 31, wv = tid >> 5;
  const int hi = lane >> 4, lo = lane & 15;
  const int mrow = (wv & 3) * 16, nh = wv >> 2;

  const __bf16* cur = sA;
  __bf16* nxt = sB;
  for (int l = 0; l < 4; ++l) {
    v8f acc[4];
#pragma unroll
    for (int i = 0; i < 4; ++i) acc[i] = vzero8();
    wmma_stripe(cur, 136, fcWf + l * 16384, 4, mrow, nh, lane, acc);
    const float* bias = (l == 0) ? b0 : (l == 1) ? b1 : (l == 2) ? b2 : b3;
#pragma unroll
    for (int nt = 0; nt < 4; ++nt) {
      int col = nh * 64 + nt * 16 + lo;
      float bb = bias[col];
#pragma unroll
      for (int v = 0; v < 8; ++v) {
        int row = mrow + v + hi * 8;
        float val = acc[nt][v] + bb;
        if (l < 3) {
          val = (val >= 0.f) ? val : 0.2f * val;       // leaky relu
          nxt[row * 136 + col] = (__bf16)val;
        } else {                                       // final layer: no act
          baseB[(long)(n0 + row) * 160 + col] = (__bf16)val;
        }
      }
    }
    __syncthreads();
    __bf16* t2 = (__bf16*)cur; cur = nxt; nxt = t2;
  }
}

// ---------------------------------------------------------------------------
// g network: one block = (64 samples, one d). Forward 4 layers + residual,
// then analytic backward for the Jacobian diagonal, all in LDS (~58 KB).
__global__ __launch_bounds__(256) void g_net_kernel(
    const float* __restrict__ x, const __bf16* __restrict__ baseB,
    const __bf16* __restrict__ g0f, const __bf16* __restrict__ g1f,
    const __bf16* __restrict__ g2f, const __bf16* __restrict__ g1tf,
    const __bf16* __restrict__ g2tf,
    const float* __restrict__ w0last, const float* __restrict__ w3v,
    const float* __restrict__ gb0, const float* __restrict__ gb1,
    const float* __restrict__ gb2, const float* __restrict__ gb3,
    float* __restrict__ outResid, float* __restrict__ outLog)
{
  __shared__ __attribute__((aligned(16))) __bf16 R0[64 * 168]; // inp -> u1
  __shared__ __attribute__((aligned(16))) __bf16 R1[64 * 136]; // h0
  __shared__ __attribute__((aligned(16))) __bf16 R2[64 * 136]; // h2 -> u2
  __shared__ float sXt[64];
  __shared__ float sDJ[64];
  __shared__ float sRed[256];

  const int tid = threadIdx.x;
  const int n0  = blockIdx.x * 64;
  const int d   = blockIdx.y;

  // warm L2/WGP$ for this d's weight fragments (global_prefetch_b8)
  __builtin_prefetch(g1f + d * 16384, 0, 3);
  __builtin_prefetch(g2f + d * 16384, 0, 3);
  __builtin_prefetch(g2tf + d * 16384, 0, 3);
  __builtin_prefetch(g1tf + d * 16384, 0, 3);

  // stage base tile (160 bf16/row) into R0
#pragma unroll
  for (int i = 0; i < 5; ++i) {
    int idx = tid + i * 256;               // 0..1279 float4s
    int row = idx / 20; int c = idx % 20;
    *(float4*)(R0 + row * 168 + c * 8) =
        *(const float4*)(baseB + (long)(n0 + row) * 160 + c * 8);
  }
  if (tid < 64) {
    int n = n0 + tid; int b = n >> 9; int t = n & 511;
    sXt[tid] = x[(long)(b * 514 + t + 2) * 16 + d];  // x_t[:,d]
    sDJ[tid] = 0.f;
  }
  __syncthreads();

  const int lane = tid & 31, wv = tid >> 5;
  const int hi = lane >> 4, lo = lane & 15;
  const int mrow = (wv & 3) * 16, nh = wv >> 2;
  const float* wl = w0last + d * 128;
  const float* w3 = w3v + d * 128;

  // ---- layer 0: K=160 GEMM + rank-1 x_t * W0[160,:] ----
  {
    v8f acc[4];
#pragma unroll
    for (int i = 0; i < 4; ++i) acc[i] = vzero8();
    wmma_stripe(R0, 168, g0f + (long)d * 20480, 5, mrow, nh, lane, acc);
#pragma unroll
    for (int nt = 0; nt < 4; ++nt) {
      int col = nh * 64 + nt * 16 + lo;
      float wlc = wl[col];
      float bb  = gb0[d * 128 + col];
#pragma unroll
      for (int v = 0; v < 8; ++v) {
        int row = mrow + v + hi * 8;
        float val = acc[nt][v] + sXt[row] * wlc + bb;
        val = (val >= 0.f) ? val : 0.2f * val;
        R1[row * 136 + col] = (__bf16)val;             // h0
      }
    }
    __syncthreads();
  }
  // ---- layer 1: h1 -> R0 (reuse, stride 168) ----
  {
    v8f acc[4];
#pragma unroll
    for (int i = 0; i < 4; ++i) acc[i] = vzero8();
    wmma_stripe(R1, 136, g1f + (long)d * 16384, 4, mrow, nh, lane, acc);
#pragma unroll
    for (int nt = 0; nt < 4; ++nt) {
      int col = nh * 64 + nt * 16 + lo;
      float bb = gb1[d * 128 + col];
#pragma unroll
      for (int v = 0; v < 8; ++v) {
        int row = mrow + v + hi * 8;
        float val = acc[nt][v] + bb;
        val = (val >= 0.f) ? val : 0.2f * val;
        R0[row * 168 + col] = (__bf16)val;             // h1
      }
    }
    __syncthreads();
  }
  // ---- layer 2: h2 -> R2 ----
  {
    v8f acc[4];
#pragma unroll
    for (int i = 0; i < 4; ++i) acc[i] = vzero8();
    wmma_stripe(R0, 168, g2f + (long)d * 16384, 4, mrow, nh, lane, acc);
#pragma unroll
    for (int nt = 0; nt < 4; ++nt) {
      int col = nh * 64 + nt * 16 + lo;
      float bb = gb2[d * 128 + col];
#pragma unroll
      for (int v = 0; v < 8; ++v) {
        int row = mrow + v + hi * 8;
        float val = acc[nt][v] + bb;
        val = (val >= 0.f) ? val : 0.2f * val;
        R2[row * 136 + col] = (__bf16)val;             // h2
      }
    }
    __syncthreads();
  }
  // ---- residual = h2.w3 + b3 ; in-place h2 -> u2 = lrelu'(h2)*w3 ----
  {
    int row = tid >> 2, q = tid & 3;
    float s = 0.f;
    for (int c = q * 32; c < q * 32 + 32; ++c) {
      float h  = (float)R2[row * 136 + c];
      float wc = w3[c];
      s += h * wc;
      R2[row * 136 + c] = (__bf16)(((h >= 0.f) ? 1.f : 0.2f) * wc);
    }
    sRed[row * 4 + q] = s;
    __syncthreads();
    if (q == 0) {
      float r = sRed[row * 4] + sRed[row * 4 + 1] + sRed[row * 4 + 2] +
                sRed[row * 4 + 3] + gb3[d];
      outResid[(long)(n0 + row) * 16 + d] = r;
    }
  }
  // ---- backward 1: v1 = u2 @ W2^T ; in-place h1 -> u1 = lrelu'(h1)*v1 ----
  {
    v8f acc[4];
#pragma unroll
    for (int i = 0; i < 4; ++i) acc[i] = vzero8();
    wmma_stripe(R2, 136, g2tf + (long)d * 16384, 4, mrow, nh, lane, acc);
#pragma unroll
    for (int nt = 0; nt < 4; ++nt) {
      int col = nh * 64 + nt * 16 + lo;
#pragma unroll
      for (int v = 0; v < 8; ++v) {
        int row = mrow + v + hi * 8;
        float h1v = (float)R0[row * 168 + col];
        float m = (h1v >= 0.f) ? 1.f : 0.2f;
        R0[row * 168 + col] = (__bf16)(m * acc[nt][v]);  // u1
      }
    }
    __syncthreads();
  }
  // ---- backward 2: v0 = u1 @ W1^T ; dJ = (lrelu'(h0)*v0) . W0[160,:] ----
  {
    v8f acc[4];
#pragma unroll
    for (int i = 0; i < 4; ++i) acc[i] = vzero8();
    wmma_stripe(R0, 168, g1tf + (long)d * 16384, 4, mrow, nh, lane, acc);
    float rsum[8] = {0.f, 0.f, 0.f, 0.f, 0.f, 0.f, 0.f, 0.f};
#pragma unroll
    for (int nt = 0; nt < 4; ++nt) {
      int col = nh * 64 + nt * 16 + lo;
      float wlc = wl[col];
#pragma unroll
      for (int v = 0; v < 8; ++v) {
        int row = mrow + v + hi * 8;
        float h0v = (float)R1[row * 136 + col];
        float m = (h0v >= 0.f) ? 1.f : 0.2f;
        rsum[v] += m * acc[nt][v] * wlc;
      }
    }
#pragma unroll
    for (int v = 0; v < 8; ++v) {
      int row = mrow + v + hi * 8;
      __hip_atomic_fetch_add(&sDJ[row], rsum[v], __ATOMIC_RELAXED,
                             __HIP_MEMORY_SCOPE_WORKGROUP);
    }
    __syncthreads();
    if (tid < 64) {
      float dj = sDJ[tid];
      __hip_atomic_fetch_add(&outLog[n0 + tid], logf(fabsf(dj)),
                             __ATOMIC_RELAXED, __HIP_MEMORY_SCOPE_AGENT);
    }
  }
}

// ---------------------------------------------------------------------------
extern "C" void kernel_launch(void* const* d_in, const int* in_sizes, int n_in,
                              void* d_out, int out_size, void* d_ws,
                              size_t ws_size, hipStream_t stream)
{
  const float* x    = (const float*)d_in[0];
  const float* emb  = (const float*)d_in[1];
  const float* fcW0 = (const float*)d_in[2];
  const float* fcB0 = (const float*)d_in[3];
  const float* fcW1 = (const float*)d_in[4];
  const float* fcB1 = (const float*)d_in[5];
  const float* fcW2 = (const float*)d_in[6];
  const float* fcB2 = (const float*)d_in[7];
  const float* fcW3 = (const float*)d_in[8];
  const float* fcB3 = (const float*)d_in[9];
  const float* gW0  = (const float*)d_in[10];
  const float* gb0  = (const float*)d_in[11];
  const float* gW1  = (const float*)d_in[12];
  const float* gb1  = (const float*)d_in[13];
  const float* gW2  = (const float*)d_in[14];
  const float* gb2  = (const float*)d_in[15];
  const float* gW3  = (const float*)d_in[16];
  const float* gb3  = (const float*)d_in[17];

  char* ws = (char*)d_ws;
  __bf16* fcWf   = (__bf16*)(ws + OFF_FCW);
  __bf16* g0f    = (__bf16*)(ws + OFF_G0F);
  __bf16* g1f    = (__bf16*)(ws + OFF_G1F);
  __bf16* g2f    = (__bf16*)(ws + OFF_G2F);
  __bf16* g1tf   = (__bf16*)(ws + OFF_G1T);
  __bf16* g2tf   = (__bf16*)(ws + OFF_G2T);
  float*  w0last = (float*)(ws + OFF_W0L);
  float*  w3v    = (float*)(ws + OFF_W3);
  __bf16* baseB  = (__bf16*)(ws + OFF_BASE);

  float* outResid = (float*)d_out;             // (B, length, D) = 131072
  float* outLog   = outResid + 131072;         // (B, length)    = 8192

  prep_kernel<<<6704, 256, 0, stream>>>(
      x, fcW0, fcW1, fcW2, fcW3, gW0, gW1, gW2, gW3,
      fcWf, g0f, g1f, g2f, g1tf, g2tf, w0last, w3v, baseB, outLog);

  fc_mlp_kernel<<<128, 256, 0, stream>>>(
      emb, fcWf, fcB0, fcB1, fcB2, fcB3, baseB);

  dim3 gg(128, 16);
  g_net_kernel<<<gg, 256, 0, stream>>>(
      x, baseB, g0f, g1f, g2f, g1tf, g2tf, w0last, w3v,
      gb0, gb1, gb2, gb3, outResid, outLog);
}
